// full_attention_7533372638049
// MI455X (gfx1250) — compile-verified, matrix operand order checked by reading, not run on hardware
//
#include <hip/hip_runtime.h>
#include <hip/hip_bf16.h>
#include <math.h>

// ---- problem constants (from setup_inputs) ----
constexpr int  Bn   = 16;     // batch
constexpr int  Dn   = 512;    // channels
constexpr int  Nn   = 2048;   // points
constexpr int  LNn  = 512;    // low-res points
constexpr int  Kk   = 16;     // TOP_K
constexpr float BN_EPS = 1e-5f;

typedef __attribute__((ext_vector_type(16))) __bf16 v16bf;
typedef __attribute__((ext_vector_type(8)))  __bf16 v8bf;
typedef __attribute__((ext_vector_type(8)))  float  v8f;
typedef __attribute__((ext_vector_type(4)))  unsigned int v4u;
typedef __attribute__((ext_vector_type(8)))  int v8i;
typedef __attribute__((ext_vector_type(4)))  int v4i;

constexpr int BM  = 32;     // block rows
constexpr int BNC = 128;    // block cols
constexpr int BK  = 32;     // K step

// ---------------------------------------------------------------------------
// Tensor Data Mover: async DMA of a 2-D fp32 tile (BK x BM, dim0 contiguous)
// from global memory into LDS. D# layout per CDNA5 ISA ch.8 (group0 128b,
// group1 256b; groups 2/3 zero for 2-D tensors; workgroup_mask=0: no cluster).
// This toolchain exposes the 6-arg builtin:
//   (u32x4 g0, i32x8 g1, i32x4 g2, i32x4 g3, i32x8 extra, i32 cpol)
// ---------------------------------------------------------------------------
__device__ __forceinline__ void tdm_load_tile_2d(unsigned int lds_off,
                                                 const float* gptr,
                                                 int lda, int K, int M)
{
    const unsigned long long ga = (unsigned long long)(uintptr_t)gptr;
    v4u g0;
    g0[0] = 1u;                                            // count=1, user desc
    g0[1] = lds_off;                                       // lds_addr (bytes)
    g0[2] = (unsigned int)(ga & 0xFFFFFFFFull);            // global_addr[31:0]
    g0[3] = (unsigned int)((ga >> 32) & 0x01FFFFFFull)     // global_addr[56:32]
          | (2u << 30);                                    // type = 2 ("image")
    v8i g1;
    g1[0] = (int)(2u << 16);                               // data_size=2 (4B)
    g1[1] = (int)(((unsigned int)K & 0xFFFFu) << 16);      // tensor_dim0[15:0]
    g1[2] = (int)((((unsigned int)K >> 16) & 0xFFFFu)      // tensor_dim0[31:16]
          | (((unsigned int)M & 0xFFFFu) << 16));          // tensor_dim1[15:0]
    g1[3] = (int)((((unsigned int)M >> 16) & 0xFFFFu)      // tensor_dim1[31:16]
          | ((unsigned int)BK << 16));                     // tile_dim0 = BK
    g1[4] = (int)BM;                                       // tile_dim1 = BM (tile_dim2=0)
    g1[5] = lda;                                           // tensor_dim0_stride[31:0]
    g1[6] = 0;                                             // stride[47:32] | dim1_stride lo
    g1[7] = 0;
    const v4i z4 = {};
    const v8i z8 = {};
    __builtin_amdgcn_tensor_load_to_lds(g0, g1, z4, z4, z8, 0);
}

// ---------------------------------------------------------------------------
// Block-tiled bf16 WMMA GEMM:  C[M x Ncols] (+bias) = A[M x K] * B[K x Ncols]
// (B optionally B0 - B1, fused on the VALU staging path).
// Block: 128 threads (4 waves) -> 32 x 128 macro-tile; wave: 2x2 WMMA tiles.
// A tile: TDM tensor_load_to_lds (raw fp32, issued by wave 0, TENSORcnt).
// B tile: cooperative load + fp32->bf16 convert + ds_store_b128 ([col][k]).
// Fragments: two aligned b128 LDS reads per B fragment; A read as fp32, cvt to bf16.
// ---------------------------------------------------------------------------
template<bool SUB, bool HAS_BIAS>
__global__ __launch_bounds__(128)
void gemm_bf16_kernel(const float* __restrict__ A, int lda, int M,
                      const float* __restrict__ B0, const float* __restrict__ B1, int ldb,
                      float* __restrict__ C, int ldc,
                      const float* __restrict__ bias, int K,
                      size_t strideB, size_t strideC)
{
    const int tid  = threadIdx.x;
    const int wid  = tid >> 5;
    const int lane = tid & 31;
    const int half = lane >> 4;
    const int mr   = lane & 15;

    const int m0   = blockIdx.y * BM;
    const int col0 = blockIdx.x * BNC;
    const int b    = blockIdx.z;

    const float* Bb0 = B0 + (size_t)b * strideB;
    const float* Bb1 = SUB ? (B1 + (size_t)b * strideB) : nullptr;
    float*       Cb  = C + (size_t)b * strideC;

    __shared__ __align__(16) float  sA[BM * BK];    // fp32 [row][k] (TDM dest)
    __shared__ __align__(16) __bf16 sB[BNC * BK];   // bf16 [col][k]

    // LDS byte offset of sA: low 32 bits of the flat address (LDS aperture
    // maps addr[31:0] directly to the LDS offset).
    const unsigned int sA_off = (unsigned int)(uintptr_t)(&sA[0]);

    v8f acc[2][2] = {};
    const int bc = tid;     // B-fill: one column per thread

    for (int k0 = 0; k0 < K; k0 += BK) {
        // ---- kick off TDM for the A tile (wave 0 only; EXEC ignored) ----
        if (wid == 0) {
            tdm_load_tile_2d(sA_off, A + (size_t)m0 * (size_t)lda + k0, lda, K, M);
        }
        // prefetch next strided B chunk
        if (k0 + BK < K) {
            __builtin_prefetch(Bb0 + (size_t)(k0 + BK) * (size_t)ldb + col0 + bc, 0, 1);
        }
        // ---- stage B tile on the VALU path (fused residual + bf16 cvt) ----
        {
            const float* s0 = Bb0 + (size_t)k0 * (size_t)ldb + col0 + bc;
            const float* s1 = SUB ? (Bb1 + (size_t)k0 * (size_t)ldb + col0 + bc) : nullptr;
            v8bf tq[4];
#pragma unroll
            for (int kk = 0; kk < BK; ++kk) {
                float v = s0[(size_t)kk * (size_t)ldb];
                if (SUB) v -= s1[(size_t)kk * (size_t)ldb];
                tq[kk >> 3][kk & 7] = (__bf16)v;
            }
#pragma unroll
            for (int q = 0; q < 4; ++q) ((v8bf*)&sB[bc * BK])[q] = tq[q];
        }
        // ---- wait for the DMA, then make both tiles visible block-wide ----
        if (wid == 0) {
            __builtin_amdgcn_s_wait_tensorcnt(0);
        }
        __syncthreads();

        // ---- assemble fragments from LDS ----
        v16bf afrag[2], bfrag[2];
#pragma unroll
        for (int rt = 0; rt < 2; ++rt) {
            const float* pa = &sA[(rt * 16 + mr) * BK + half * 8];
            v16bf af;
#pragma unroll
            for (int j = 0; j < 8; ++j) {
                af[j]     = (__bf16)pa[j];        // K = half*8 + j
                af[j + 8] = (__bf16)pa[j + 16];   // K = 16 + half*8 + j
            }
            afrag[rt] = af;
        }
#pragma unroll
        for (int ct = 0; ct < 2; ++ct) {
            const __bf16* pb = &sB[(wid * 32 + ct * 16 + mr) * BK + half * 8];
            v8bf lo = *(const v8bf*)pb;
            v8bf hi = *(const v8bf*)(pb + 16);
            bfrag[ct] = __builtin_shufflevector(lo, hi, 0,1,2,3,4,5,6,7,
                                                        8,9,10,11,12,13,14,15);
        }
        // ---- 4 WMMAs per K-step ----
#pragma unroll
        for (int rt = 0; rt < 2; ++rt)
#pragma unroll
            for (int ct = 0; ct < 2; ++ct)
                acc[rt][ct] = __builtin_amdgcn_wmma_f32_16x16x32_bf16(
                    false, afrag[rt], false, bfrag[ct],
                    (short)0, acc[rt][ct], false, false);
        __syncthreads();   // tiles free for next iteration's fill/DMA
    }

    // ---- epilogue (bias compile-time) ----
#pragma unroll
    for (int rt = 0; rt < 2; ++rt) {
#pragma unroll
        for (int r = 0; r < 8; ++r) {
            const int row = m0 + rt * 16 + r + half * 8;
            float bv = 0.0f;
            if (HAS_BIAS) bv = bias[row];
#pragma unroll
            for (int ct = 0; ct < 2; ++ct) {
                const int col = col0 + wid * 32 + ct * 16 + mr;
                Cb[(size_t)row * (size_t)ldc + col] = acc[rt][ct][r] + bv;
            }
        }
    }
}

// ---- top-16 per row of low_attention (strict > => ties keep lower index) ----
__global__ void topk_kernel(const float* __restrict__ low,
                            float* __restrict__ tvals, int* __restrict__ tidx)
{
    const int t = blockIdx.x * blockDim.x + threadIdx.x;   // over B*LN rows
    if (t >= Bn * LNn) return;
    const float* row = low + (size_t)t * LNn;
    float v[Kk];
    int   ix[Kk];
#pragma unroll
    for (int j = 0; j < Kk; ++j) { v[j] = -INFINITY; ix[j] = 0; }
    for (int c = 0; c < LNn; ++c) {
        const float xv = row[c];
        if (xv > v[Kk - 1]) {
            int p = Kk - 1;
            while (p > 0 && xv > v[p - 1]) { v[p] = v[p - 1]; ix[p] = ix[p - 1]; --p; }
            v[p] = xv; ix[p] = c;
        }
    }
#pragma unroll
    for (int j = 0; j < Kk; ++j) {
        tvals[(size_t)t * Kk + j] = v[j];
        tidx [(size_t)t * Kk + j] = ix[j];
    }
}

__global__ void zero_kernel(float* __restrict__ p, size_t n)
{
    const size_t t = (size_t)blockIdx.x * blockDim.x + threadIdx.x;
    if (t < n) p[t] = 0.0f;
}

// ---- scatter top-k into dense att for one batch ----
__global__ void scatter_kernel(const float* __restrict__ tvals, const int* __restrict__ tidx,
                               const int* __restrict__ fps, const int* __restrict__ knn,
                               float* __restrict__ att, int b)
{
    const int t = blockIdx.x * blockDim.x + threadIdx.x;   // over LN*K
    if (t >= LNn * Kk) return;
    const int i = t / Kk;
    const int j = t % Kk;
    const size_t base = ((size_t)b * LNn + i) * Kk + j;
    const int   ti  = tidx[base];
    const float val = tvals[base];
    const int row = fps[(size_t)b * LNn + i];
    const int c1  = fps[(size_t)b * LNn + ti];
    const int c2  = knn[((size_t)b * LNn + ti) * Kk + j];
    att[(size_t)row * Nn + c1] = val;   // gti scatter
    att[(size_t)row * Nn + c2] = val;   // knn scatter (later write wins in-thread)
}

// ---- BatchNorm stats: one block per channel d, reduce over B*N ----
__global__ void bn_stats_kernel(const float* __restrict__ h,
                                float* __restrict__ mean, float* __restrict__ rstd)
{
    const int d = blockIdx.x;
    float s = 0.0f, s2 = 0.0f;
    for (int t = threadIdx.x; t < Bn * Nn; t += blockDim.x) {
        const int b = t >> 11;          // / Nn
        const int n = t & (Nn - 1);
        const float v = h[((size_t)b * Dn + d) * Nn + n];
        s += v; s2 += v * v;
    }
    __shared__ float sh[256], sh2[256];
    sh[threadIdx.x] = s; sh2[threadIdx.x] = s2;
    __syncthreads();
    for (int off = 128; off > 0; off >>= 1) {
        if (threadIdx.x < off) {
            sh [threadIdx.x] += sh [threadIdx.x + off];
            sh2[threadIdx.x] += sh2[threadIdx.x + off];
        }
        __syncthreads();
    }
    if (threadIdx.x == 0) {
        const float inv = 1.0f / (float)(Bn * Nn);
        const float m   = sh[0] * inv;
        const float var = sh2[0] * inv - m * m;
        mean[d] = m;
        rstd[d] = rsqrtf(var + BN_EPS);
    }
}

// ---- out = x + relu(gamma*(h-mean)*rstd + beta) ----
__global__ void bn_relu_res_kernel(const float* __restrict__ x, const float* __restrict__ h,
                                   const float* __restrict__ mean, const float* __restrict__ rstd,
                                   const float* __restrict__ gamma, const float* __restrict__ beta,
                                   float* __restrict__ out)
{
    const size_t t = (size_t)blockIdx.x * blockDim.x + threadIdx.x;
    if (t >= (size_t)Bn * Dn * Nn) return;
    const int d = (int)((t >> 11) & (Dn - 1));   // (t / Nn) % Dn
    float v = (h[t] - mean[d]) * rstd[d] * gamma[d] + beta[d];
    v = v > 0.0f ? v : 0.0f;
    out[t] = x[t] + v;
}

// ---------------------------------------------------------------------------
extern "C" void kernel_launch(void* const* d_in, const int* in_sizes, int n_in,
                              void* d_out, int out_size, void* d_ws, size_t ws_size,
                              hipStream_t stream)
{
    (void)in_sizes; (void)n_in; (void)out_size; (void)ws_size;

    const float* x     = (const float*)d_in[0];
    const float* low   = (const float*)d_in[1];
    const int*   fps   = (const int*)  d_in[2];
    const int*   knn   = (const int*)  d_in[3];
    const float* vw    = (const float*)d_in[4];
    const float* vb    = (const float*)d_in[5];
    const float* tw    = (const float*)d_in[6];
    const float* tb    = (const float*)d_in[7];
    const float* gamma = (const float*)d_in[8];
    const float* beta  = (const float*)d_in[9];
    float*       out   = (float*)d_out;

    // workspace carve-up
    float* xv    = (float*)d_ws;                           // B*D*N  (reused as h)
    float* xr    = xv + (size_t)Bn * Dn * Nn;              // B*D*N
    float* att   = xr + (size_t)Bn * Dn * Nn;              // N*N (per-batch, reused)
    float* tvals = att + (size_t)Nn * Nn;                  // B*LN*K
    int*   tidx  = (int*)(tvals + (size_t)Bn * LNn * Kk);  // B*LN*K
    float* mean  = (float*)(tidx + (size_t)Bn * LNn * Kk); // D
    float* rstd  = mean + Dn;                              // D

    const size_t DN = (size_t)Dn * Nn;
    const dim3 grid_full(Nn / BNC, Dn / BM, Bn);   // 16 x 16 x 16
    const dim3 grid_one (Nn / BNC, Dn / BM, 1);    // per-batch att GEMM
    const int  blk = 128;

    // 1) x_v = v_w @ x + v_b
    gemm_bf16_kernel<false, true><<<grid_full, blk, 0, stream>>>(
        vw, Dn, Dn, x, nullptr, Nn, xv, Nn, vb, Dn, DN, DN);

    // 2) top-k of low_attention rows
    topk_kernel<<<(Bn * LNn + 255) / 256, 256, 0, stream>>>(low, tvals, tidx);

    // 3) per batch: zero att -> scatter -> x_r[b] = x_v[b] @ att
    const size_t attN = (size_t)Nn * Nn;
    for (int b = 0; b < Bn; ++b) {
        zero_kernel<<<(unsigned)((attN + 255) / 256), 256, 0, stream>>>(att, attN);
        scatter_kernel<<<(LNn * Kk + 255) / 256, 256, 0, stream>>>(tvals, tidx, fps, knn, att, b);
        gemm_bf16_kernel<false, false><<<grid_one, blk, 0, stream>>>(
            xv + (size_t)b * DN, Nn, Dn, att, nullptr, Nn,
            xr + (size_t)b * DN, Nn, nullptr, Nn, 0, 0);
    }

    // 4) h = t_w @ (x - x_r) + t_b   (h overwrites xv buffer)
    float* h = xv;
    gemm_bf16_kernel<true, true><<<grid_full, blk, 0, stream>>>(
        tw, Dn, Dn, x, xr, Nn, h, Nn, tb, Dn, DN, DN);

    // 5) BN stats per channel
    bn_stats_kernel<<<Dn, 256, 0, stream>>>(h, mean, rstd);

    // 6) out = x + relu(bn(h))
    const size_t tot = (size_t)Bn * Dn * Nn;
    bn_relu_res_kernel<<<(unsigned)((tot + 255) / 256), 256, 0, stream>>>(
        x, h, mean, rstd, gamma, beta, out);
}
